// MultiHeadAttention_6725918785726
// MI455X (gfx1250) — matrix operand order repaired and checked by reading, not compiled, not run on hardware
//
#include <hip/hip_runtime.h>

typedef __bf16 bhalf;
typedef __attribute__((ext_vector_type(16))) __bf16 v16bf;
typedef __attribute__((ext_vector_type(8)))  float  v8f;

#define EMBED 1024
#define NH    16
#define HD    64
#define SEQ   2048
#define BATCH 4

// ---------------------------------------------------------------------------
// VALU xor-shuffle within 16-lane rows via v_permlane16_b32 (no DS, no waits)
// ---------------------------------------------------------------------------
template <int M>
__device__ __forceinline__ float permxor16(float v) {
  constexpr unsigned s0 = (M == 1) ? 0x67452301u
                        : (M == 2) ? 0x54761032u
                        : (M == 4) ? 0x32107654u
                                   : 0xFEDCBA98u;   // M == 8
  constexpr unsigned s1 = (M == 1) ? 0xEFCDAB89u
                        : (M == 2) ? 0xDCFE98BAu
                        : (M == 4) ? 0xBA98FEDCu
                                   : 0x76543210u;   // M == 8
  unsigned u = __float_as_uint(v);
  unsigned r = __builtin_amdgcn_permlane16(u, u, s0, s1, false, false);
  return __uint_as_float(r);
}

// ---------------------------------------------------------------------------
// f32 -> bf16 conversion (row-major copy, for the A matrix x)
// ---------------------------------------------------------------------------
__global__ void cvt_f32_bf16(const float* __restrict__ src,
                             bhalf* __restrict__ dst, int n) {
  int i = blockIdx.x * blockDim.x + threadIdx.x;
  if (i < n) dst[i] = (bhalf)src[i];
}

// ---------------------------------------------------------------------------
// f32 [K x Nc] -> bf16 [Nc x K] tiled transpose (for weight matrices)
// block (32,8), grid (Nc/32, K/32)
// ---------------------------------------------------------------------------
__global__ void cvt_transpose_bf16(const float* __restrict__ src,
                                   bhalf* __restrict__ dst, int K, int Nc) {
  __shared__ bhalf tile[32][33];
  const int bx = blockIdx.x * 32;   // Nc offset
  const int by = blockIdx.y * 32;   // K offset
  const int tx = threadIdx.x;       // 0..31
  const int ty = threadIdx.y;       // 0..7
#pragma unroll
  for (int r = 0; r < 32; r += 8)
    tile[ty + r][tx] = (bhalf)src[(size_t)(by + ty + r) * Nc + bx + tx];
  __syncthreads();
#pragma unroll
  for (int r = 0; r < 32; r += 8)
    dst[(size_t)(bx + ty + r) * K + by + tx] = tile[tx][ty + r];
}

// ---------------------------------------------------------------------------
// Tiled bf16 WMMA GEMM with TRANSPOSED B:  C[M x Nc] = A[M x K] * Bt^T + bias
//   A  : bf16 [M x K]  row-major
//   Bt : bf16 [Nc x K] (column-major B) -> every B fragment is one 32B load
// Block = 256 threads = 8 waves; block tile 64(M) x 256(N);
// each wave computes 32 x 64 (2x4 WMMA accumulators, 8 wmma per k-step).
// MODE 0: scatter bf16 into q,k (head-major [BH][SEQ][HD]) and vT ([BH][HD][SEQ])
// MODE 1: f32 store to Out[M x Nc]
// ---------------------------------------------------------------------------
template <int MODE>
__global__ void gemm_bf16(const bhalf* __restrict__ A,
                          const bhalf* __restrict__ Bt,
                          const float* __restrict__ bias,
                          void* __restrict__ Out,
                          int K, int Nc) {
  const int lane = threadIdx.x & 31;
  const int wid  = threadIdx.x >> 5;              // 0..7
  const int g    = lane >> 4;                     // 16-lane group
  const int l15  = lane & 15;
  const int m0   = blockIdx.y * 64  + (wid & 1) * 32;
  const int n0   = blockIdx.x * 256 + (wid >> 1) * 64;

  v8f acc[2][4] = {{v8f{}, v8f{}, v8f{}, v8f{}}, {v8f{}, v8f{}, v8f{}, v8f{}}};

  for (int k0 = 0; k0 < K; k0 += 32) {
    // A fragments: lane holds row m0+r*16+l15; elem j <-> K = k0+g*16+j
    v16bf a[2];
#pragma unroll
    for (int r = 0; r < 2; ++r) {
      const bhalf* ar = A + (size_t)(m0 + r * 16 + l15) * K + k0 + g * 16;
#pragma unroll
      for (int j = 0; j < 8; ++j) { a[r][j] = ar[j]; a[r][8 + j] = ar[8 + j]; }
    }
#pragma unroll
    for (int t = 0; t < 4; ++t) {
      // B fragment: col n0+t*16+l15 ; elem j <-> K = k0+g*16+j  (contiguous 32B)
      const bhalf* br = Bt + (size_t)(n0 + t * 16 + l15) * K + k0 + g * 16;
      __builtin_prefetch(br + 32, 0, 1);
      const v16bf b = *(const v16bf*)br;
#pragma unroll
      for (int r = 0; r < 2; ++r)
        acc[r][t] = __builtin_amdgcn_wmma_f32_16x16x32_bf16(
            false, a[r], false, b, (short)0, acc[r][t], false, false);
    }
  }

#pragma unroll
  for (int r = 0; r < 2; ++r) {
#pragma unroll
    for (int t = 0; t < 4; ++t) {
#pragma unroll
      for (int i = 0; i < 8; ++i) {
        const int row = m0 + r * 16 + i + 8 * g;   // C layout: row = i + 8*g
        const int col = n0 + t * 16 + l15;         //           col = lane&15
        const float v = acc[r][t][i] + bias[col];
        if (MODE == 0) {
          bhalf* outb = (bhalf*)Out;               // base of q|k|vT region
          const int which = col >> 10;             // 0=q 1=k 2=v
          const int h     = (col & 1023) >> 6;
          const int hd    = col & 63;
          const int b     = row >> 11;
          const int n     = row & 2047;
          const size_t bh   = (size_t)b * NH + h;
          const size_t span = (size_t)BATCH * NH * SEQ * HD;
          if (which == 2) {                        // V stored transposed per head
            outb[2 * span + (bh * HD + hd) * SEQ + n] = (bhalf)v;
          } else {                                 // Q, K head-major row-major
            outb[(size_t)which * span + (bh * SEQ + n) * HD + hd] = (bhalf)v;
          }
        } else {
          ((float*)Out)[(size_t)row * Nc + col] = v;
        }
      }
    }
  }
}

// ---------------------------------------------------------------------------
// Flash attention: one wave per 16-query tile, 4 waves per block.
// Base-2 online softmax; cross-lane reductions via v_permlane16_b32.
// K is [BH][SEQ][HD] (score B-frags contiguous); V is [BH][HD][SEQ]
// (PV B-frags contiguous). P re-striped C-layout -> A-layout through LDS.
// ---------------------------------------------------------------------------
__global__ void attn_kernel(const bhalf* __restrict__ qb,
                            const bhalf* __restrict__ kb,
                            const bhalf* __restrict__ vtb,
                            bhalf* __restrict__ outb) {
  __shared__ float sld[4][16][32];
  const int lane = threadIdx.x & 31;
  const int wid  = threadIdx.x >> 5;              // 0..3
  const int g    = lane >> 4;
  const int l15  = lane & 15;
  const int bh   = blockIdx.y;                    // 0..63
  const int h    = bh & 15;
  const int b    = bh >> 4;
  const int q0   = blockIdx.x * 64 + wid * 16;

  const size_t head = (size_t)bh * SEQ * HD;      // same total size for vT

  // Q fragments for head-dim chunks [0,32) and [32,64)
  v16bf aq0, aq1;
  {
    const bhalf* qr = qb + head + (size_t)(q0 + l15) * HD;
#pragma unroll
    for (int j = 0; j < 8; ++j) {
      aq0[j]     = qr[g * 16 + j];
      aq0[8 + j] = qr[g * 16 + 8 + j];
      aq1[j]     = qr[32 + g * 16 + j];
      aq1[8 + j] = qr[32 + g * 16 + 8 + j];
    }
  }

  v8f o[4] = {v8f{}, v8f{}, v8f{}, v8f{}};
  float mx[8], ls[8];
#pragma unroll
  for (int i = 0; i < 8; ++i) { mx[i] = -1e30f; ls[i] = 0.0f; }
  // base-2 softmax: scale2 = (1/sqrt(64)) * log2(e)
  const float scale2 = 0.125f * 1.44269504088896340736f;

  for (int j0 = 0; j0 < SEQ; j0 += 32) {
    // ---- scores: S(16 x 32) = Q(16x64) . K^T, two 16x16 WMMA tiles ----
    v8f s0 = {}, s1 = {};
#pragma unroll
    for (int c = 0; c < 2; ++c) {
      const v16bf bk0 =
          *(const v16bf*)(kb + head + (size_t)(j0 + l15) * HD + c * 32 + g * 16);
      const v16bf bk1 =
          *(const v16bf*)(kb + head + (size_t)(j0 + 16 + l15) * HD + c * 32 + g * 16);
      const v16bf a = c ? aq1 : aq0;
      s0 = __builtin_amdgcn_wmma_f32_16x16x32_bf16(false, a, false, bk0,
                                                   (short)0, s0, false, false);
      s1 = __builtin_amdgcn_wmma_f32_16x16x32_bf16(false, a, false, bk1,
                                                   (short)0, s1, false, false);
    }

    // ---- base-2 online softmax, reductions batched over all 8 rows ----
    float cm[8], rs[8];
#pragma unroll
    for (int i = 0; i < 8; ++i) {
      s0[i] *= scale2;
      s1[i] *= scale2;
      cm[i] = fmaxf(s0[i], s1[i]);
    }
#pragma unroll
    for (int i = 0; i < 8; ++i) cm[i] = fmaxf(cm[i], permxor16<1>(cm[i]));
#pragma unroll
    for (int i = 0; i < 8; ++i) cm[i] = fmaxf(cm[i], permxor16<2>(cm[i]));
#pragma unroll
    for (int i = 0; i < 8; ++i) cm[i] = fmaxf(cm[i], permxor16<4>(cm[i]));
#pragma unroll
    for (int i = 0; i < 8; ++i) cm[i] = fmaxf(cm[i], permxor16<8>(cm[i]));
#pragma unroll
    for (int i = 0; i < 8; ++i) {
      const float mn    = fmaxf(mx[i], cm[i]);
      const float alpha = __builtin_amdgcn_exp2f(mx[i] - mn);
      mx[i] = mn;
      const float p0 = __builtin_amdgcn_exp2f(s0[i] - mn);
      const float p1 = __builtin_amdgcn_exp2f(s1[i] - mn);
      s0[i] = p0; s1[i] = p1;
      rs[i] = p0 + p1;
      ls[i] *= alpha;
      o[0][i] *= alpha; o[1][i] *= alpha; o[2][i] *= alpha; o[3][i] *= alpha;
    }
#pragma unroll
    for (int i = 0; i < 8; ++i) rs[i] += permxor16<1>(rs[i]);
#pragma unroll
    for (int i = 0; i < 8; ++i) rs[i] += permxor16<2>(rs[i]);
#pragma unroll
    for (int i = 0; i < 8; ++i) rs[i] += permxor16<4>(rs[i]);
#pragma unroll
    for (int i = 0; i < 8; ++i) rs[i] += permxor16<8>(rs[i]);
#pragma unroll
    for (int i = 0; i < 8; ++i) ls[i] += rs[i];

    // ---- re-stripe P from C-layout to A-layout through LDS ----
    __syncthreads();
#pragma unroll
    for (int i = 0; i < 8; ++i) {
      sld[wid][i + 8 * g][l15]      = s0[i];
      sld[wid][i + 8 * g][16 + l15] = s1[i];
    }
    __syncthreads();
    v16bf ap;
#pragma unroll
    for (int j = 0; j < 16; ++j) {
      const int kk = g * 16 + j;
      ap[j] = (bhalf)sld[wid][l15][kk];
    }

    // ---- O(16 x 64) += P(16x32) . V(32x64), V transposed: 32B B-frag loads --
#pragma unroll
    for (int t = 0; t < 4; ++t) {
      const v16bf bv = *(const v16bf*)(
          vtb + head + ((size_t)t * 16 + l15) * SEQ + j0 + g * 16);
      o[t] = __builtin_amdgcn_wmma_f32_16x16x32_bf16(false, ap, false, bv,
                                                     (short)0, o[t], false, false);
    }
  }

  // ---- finalize: divide by row sum, store bf16 [B*SEQ, EMBED] ----
#pragma unroll
  for (int i = 0; i < 8; ++i) {
    const float inv   = 1.0f / ls[i];
    const int   row   = q0 + i + 8 * g;
    const size_t base = ((size_t)b * SEQ + row) * EMBED + h * HD + l15;
    outb[base]      = (bhalf)(o[0][i] * inv);
    outb[base + 16] = (bhalf)(o[1][i] * inv);
    outb[base + 32] = (bhalf)(o[2][i] * inv);
    outb[base + 48] = (bhalf)(o[3][i] * inv);
  }
}

// ---------------------------------------------------------------------------
// Launcher
// ---------------------------------------------------------------------------
extern "C" void kernel_launch(void* const* d_in, const int* in_sizes, int n_in,
                              void* d_out, int out_size, void* d_ws, size_t ws_size,
                              hipStream_t stream) {
  (void)in_sizes; (void)n_in; (void)out_size; (void)ws_size;

  const float* x      = (const float*)d_in[0];   // [4,2048,1024]
  const float* W_qkv  = (const float*)d_in[1];   // [1024,3072]
  const float* b_qkv  = (const float*)d_in[2];   // [3072]
  const float* W_proj = (const float*)d_in[3];   // [1024,1024]
  const float* b_proj = (const float*)d_in[4];   // [1024]
  float* out          = (float*)d_out;           // [4,2048,1024]

  const int M  = BATCH * SEQ;                    // 8192
  const int nx = M * EMBED;                      // 8,388,608

  char* ws = (char*)d_ws;
  bhalf* xb     = (bhalf*)(ws);                                    // 16 MiB
  bhalf* wqkvt  = (bhalf*)(ws + (size_t)16 * 1024 * 1024);         //  6 MiB [3072 x 1024]
  bhalf* wprojt = (bhalf*)(ws + (size_t)22 * 1024 * 1024);         //  2 MiB [1024 x 1024]
  bhalf* qkvb   = (bhalf*)(ws + (size_t)24 * 1024 * 1024);         // 48 MiB (q, k, vT)
  bhalf* attnb  = (bhalf*)(ws + (size_t)72 * 1024 * 1024);         // 16 MiB

  const size_t headtot = (size_t)BATCH * NH * SEQ * HD;
  bhalf* qb  = qkvb;
  bhalf* kb  = qkvb + headtot;
  bhalf* vtb = qkvb + 2 * headtot;

  // 1) conversions: x row-major; weights transposed (bf16 [Nc x K])
  cvt_f32_bf16<<<(nx + 255) / 256, 256, 0, stream>>>(x, xb, nx);
  {
    dim3 blk(32, 8);
    dim3 gq(3 * EMBED / 32, EMBED / 32);         // (96, 32)
    cvt_transpose_bf16<<<gq, blk, 0, stream>>>(W_qkv, wqkvt, EMBED, 3 * EMBED);
    dim3 gp(EMBED / 32, EMBED / 32);             // (32, 32)
    cvt_transpose_bf16<<<gp, blk, 0, stream>>>(W_proj, wprojt, EMBED, EMBED);
  }

  // 2) QKV projection -> head-major q, k and transposed vT (bf16)
  {
    dim3 grid(3 * EMBED / 256, M / 64);          // (12, 128)
    gemm_bf16<0><<<grid, 256, 0, stream>>>(xb, wqkvt, b_qkv, (void*)qkvb,
                                           EMBED, 3 * EMBED);
  }

  // 3) Flash attention -> bf16 [8192 x 1024]
  {
    dim3 grid(SEQ / 64, BATCH * NH);             // (32, 64)
    attn_kernel<<<grid, 128, 0, stream>>>(qb, kb, vtb, attnb);
  }

  // 4) Output projection -> f32 d_out
  {
    dim3 grid(EMBED / 256, M / 64);              // (4, 128)
    gemm_bf16<1><<<grid, 256, 0, stream>>>(attnb, wprojt, b_proj, (void*)out,
                                           EMBED, EMBED);
  }
}
